// GAN_52183852647003
// MI455X (gfx1250) — matrix operand order repaired and checked by reading, not compiled, not run on hardware
//
#include <hip/hip_runtime.h>
#include <hip/hip_bf16.h>
#include <math.h>
#include <stdint.h>

// ---------------------------------------------------------------------------
// GAT + DiffPool forward for MI455X (gfx1250, wave32).
// Memory-bound: 168MB atts stores dominate -> non-temporal stores so that
// adj (33.5MB) stays resident in the 192MB L2 across 9 passes.
// WMMA (v_wmma_f32_16x16x32_f16) for the diff-pool GEMMs (K % 32 == 0).
// adj rows staged into LDS via GLOBAL_LOAD_ASYNC_TO_LDS_B128 (ASYNCcnt).
// ---------------------------------------------------------------------------

typedef __attribute__((ext_vector_type(16))) _Float16 v16h;
typedef __attribute__((ext_vector_type(8)))  float    v8f;

#define NN   2048
#define BB   2
#define ROWS (BB * NN)
#define NEG_INF_E (-9.0e15f)

// ---------------- wave32 reductions -----------------
__device__ __forceinline__ float wave_reduce_sum(float v) {
#pragma unroll
  for (int off = 16; off > 0; off >>= 1) v += __shfl_xor(v, off, 32);
  return v;
}
__device__ __forceinline__ float wave_reduce_max(float v) {
#pragma unroll
  for (int off = 16; off > 0; off >>= 1) v = fmaxf(v, __shfl_xor(v, off, 32));
  return v;
}

// ---------------- WMMA fragment loaders (ISA 7.12.2, wave32) ----------------
// A: 16x32 f16. lane<16: VGPR0..3 K=0..7, VGPR4..7 K=16..23; lane>=16: +8.
__device__ __forceinline__ v16h load_a_rowmajor(const float* base, int row0,
                                                int k0, int ld) {
  int lane = threadIdx.x & 31, half = lane >> 4, r = (lane & 15) + row0;
  const float* p = base + (size_t)r * ld + k0;
  v16h a;
#pragma unroll
  for (int e = 0; e < 8; ++e) a[e] = (_Float16)p[half * 8 + e];
#pragma unroll
  for (int e = 0; e < 8; ++e) a[8 + e] = (_Float16)p[16 + half * 8 + e];
  return a;
}
// A(m,k) = base[k*ld + m]  (source stored transposed)
__device__ __forceinline__ v16h load_a_colmajor(const float* base, int row0,
                                                int k0, int ld) {
  int lane = threadIdx.x & 31, half = lane >> 4, r = (lane & 15) + row0;
  v16h a;
#pragma unroll
  for (int e = 0; e < 8; ++e)
    a[e] = (_Float16)base[(size_t)(k0 + half * 8 + e) * ld + r];
#pragma unroll
  for (int e = 0; e < 8; ++e)
    a[8 + e] = (_Float16)base[(size_t)(k0 + 16 + half * 8 + e) * ld + r];
  return a;
}
// B: 32x16 f16, B(k,n); lane<16: elems 0..7 K=0..7, 8..15 K=16..23; lane>=16: +8.
__device__ __forceinline__ v16h load_b_rowmajor(const float* base, int k0,
                                                int col0, int ld) {
  int lane = threadIdx.x & 31, half = lane >> 4, n = (lane & 15) + col0;
  v16h b;
#pragma unroll
  for (int e = 0; e < 16; ++e)
    b[e] = (_Float16)base[(size_t)(k0 + (e < 8 ? half * 8 + e : 8 + half * 8 + e)) * ld + n];
  return b;
}
// B(k,n) = base[n*ld + k]  (source row-major over n => transposed B)
__device__ __forceinline__ v16h load_b_colmajor(const float* base, int k0,
                                                int col0, int ld) {
  int lane = threadIdx.x & 31, half = lane >> 4, n = (lane & 15) + col0;
  const float* p = base + (size_t)n * ld + k0;
  v16h b;
#pragma unroll
  for (int e = 0; e < 16; ++e) b[e] = (_Float16)p[e < 8 ? half * 8 + e : 8 + half * 8 + e];
  return b;
}

// ---------------- small utility kernels ----------------
__global__ void zero_kernel(float* p, int n) {
  int i = blockIdx.x * blockDim.x + threadIdx.x;
  if (i < n) p[i] = 0.f;
}

// out[r,o] = sum_k X[r,k] * W[k,o]
__global__ void proj_kernel(const float* __restrict__ X,
                            const float* __restrict__ W,
                            float* __restrict__ out, int rows, int fin,
                            int fout) {
  int idx = blockIdx.x * blockDim.x + threadIdx.x;
  if (idx >= rows * fout) return;
  int r = idx / fout, o = idx % fout;
  const float* xr = X + (size_t)r * fin;
  float acc = 0.f;
  for (int k = 0; k < fin; ++k) acc += xr[k] * W[k * fout + o];
  out[idx] = acc;
}

// f1[r] = Wh[r,:]·a[:fout], f2[r] = Wh[r,:]·a[fout:]
__global__ void attvec_kernel(const float* __restrict__ Wh,
                              const float* __restrict__ a,
                              float* __restrict__ f1, float* __restrict__ f2,
                              int rows, int fout) {
  int r = blockIdx.x * blockDim.x + threadIdx.x;
  if (r >= rows) return;
  const float* w = Wh + (size_t)r * fout;
  float s1 = 0.f, s2 = 0.f;
  for (int k = 0; k < fout; ++k) {
    s1 += w[k] * a[k];
    s2 += w[k] * a[fout + k];
  }
  f1[r] = s1;
  f2[r] = s2;
}

// ---------------- fused GAT attention row kernel ----------------
// One 256-thread block (8 waves) per (b,i) row. adj row staged into LDS with
// async global->LDS B128 copies (ASYNCcnt); masked leaky-relu logits; stable
// softmax in LDS; att@Wh aggregation via ds float atomics; atts emitted with
// non-temporal stores so adj stays L2-resident.
template <int HID, bool WRITE_ATT, bool ELU, bool POOL1>
__global__ __launch_bounds__(256) void gat_attn_kernel(
    const float* __restrict__ adj, const float* __restrict__ Wh,
    const float* __restrict__ f1, const float* __restrict__ f2,
    float* __restrict__ att_out, float* __restrict__ h_out, int hstride,
    int hoffset, const float* __restrict__ lw, const float* __restrict__ lb,
    float* __restrict__ s_out) {
  __shared__ float arow_s[NN];
  __shared__ float row[NN];
  __shared__ float red[8];
  __shared__ float hacc[HID];
  __shared__ float gp[32], sp[32], mm[2];

  const int tid = threadIdx.x;
  const int b = blockIdx.x / NN;
  const int i = blockIdx.x % NN;
  const size_t rbase = (size_t)b * NN + i;
  const float* arow = adj + rbase * NN;
  const float f1i = f1[b * NN + i];
  const float* f2b = f2 + b * NN;

  // async copy of this row's 8KB adjacency slice into LDS (32B per thread)
  {
    unsigned lds = (unsigned)(size_t)(&arow_s[tid * 8]);
    const float* gsrc = arow + tid * 8;
    asm volatile(
        "global_load_async_to_lds_b128 %0, %1, off\n\t"
        "global_load_async_to_lds_b128 %0, %1, off offset:16\n\t"
        "s_wait_asynccnt 0"
        :
        : "v"(lds), "v"(gsrc)
        : "memory");
  }
  __syncthreads();

  // pass 1: logits + max
  float lmax = -3.4e38f;
  for (int j = tid; j < NN; j += 256) {
    if (j + 256 < NN) __builtin_prefetch(f2b + j + 256, 0, 1);
    float e = f1i + f2b[j];
    e = (e > 0.f) ? e : 0.2f * e;
    e = (arow_s[j] > 0.f) ? e : NEG_INF_E;
    row[j] = e;
    lmax = fmaxf(lmax, e);
  }
  lmax = wave_reduce_max(lmax);
  if ((tid & 31) == 0) red[tid >> 5] = lmax;
  __syncthreads();
  float m = red[0];
#pragma unroll
  for (int w = 1; w < 8; ++w) m = fmaxf(m, red[w]);
  __syncthreads();

  // pass 2: exp + sum
  float lsum = 0.f;
  for (int j = tid; j < NN; j += 256) {
    float v = __expf(row[j] - m);
    row[j] = v;
    lsum += v;
  }
  lsum = wave_reduce_sum(lsum);
  if ((tid & 31) == 0) red[tid >> 5] = lsum;
  if (tid < HID) hacc[tid] = 0.f;
  __syncthreads();
  float s = red[0];
#pragma unroll
  for (int w = 1; w < 8; ++w) s += red[w];
  const float inv = 1.f / s;

  // pass 3: normalize, (optionally) emit att row (NT store), h = att @ Wh
  float acc[HID];
#pragma unroll
  for (int k = 0; k < HID; ++k) acc[k] = 0.f;
  for (int j = tid; j < NN; j += 256) {
    float a = row[j] * inv;
    if constexpr (WRITE_ATT)
      __builtin_nontemporal_store(a, &att_out[rbase * NN + j]);
    const float* whj = Wh + ((size_t)b * NN + j) * HID;
#pragma unroll
    for (int k = 0; k < HID; ++k) acc[k] += a * whj[k];
  }
#pragma unroll
  for (int k = 0; k < HID; ++k) atomicAdd(&hacc[k], acc[k]);
  __syncthreads();

  if constexpr (!POOL1) {
    if (tid < HID) {
      float v = hacc[tid];
      if (ELU) v = (v > 0.f) ? v : (__expf(v) - 1.f);
      h_out[((size_t)b * NN + i) * hstride + hoffset + tid] = v;
    }
  } else {
    // elu -> @lin1_w + lin1_b -> softmax over 32 clusters -> s1 row
    if (tid < 32) {
      float v = hacc[tid];
      gp[tid] = (v > 0.f) ? v : (__expf(v) - 1.f);
    }
    __syncthreads();
    if (tid < 32) {
      float v = lb[tid];
      for (int k = 0; k < 32; ++k) v += gp[k] * lw[k * 32 + tid];
      sp[tid] = v;
    }
    __syncthreads();
    if (tid == 0) {
      float mx = sp[0];
      for (int k = 1; k < 32; ++k) mx = fmaxf(mx, sp[k]);
      float sum = 0.f;
      for (int k = 0; k < 32; ++k) sum += __expf(sp[k] - mx);
      mm[0] = mx;
      mm[1] = 1.f / sum;
    }
    __syncthreads();
    if (tid < 32)
      s_out[((size_t)b * NN + i) * 32 + tid] = __expf(sp[tid] - mm[0]) * mm[1];
  }
}

// ---------------- WMMA kernels ----------------
// link1 partial: per 16x16 tile, P = s1 s1^T (one 16x16x32 WMMA), then
// sum (adj - P)^2, wave-reduce, one atomic per wave.
__global__ __launch_bounds__(256) void wmma_link1_kernel(
    const float* __restrict__ adj, const float* __restrict__ s1,
    float* __restrict__ linksum) {
  int wg = blockIdx.x * 8 + (threadIdx.x >> 5);  // 0..32767, one tile per wave
  int b = wg / 16384;
  int rem = wg % 16384;
  int it = rem >> 7, jt = rem & 127;
  const float* S = s1 + (size_t)b * NN * 32;
  v16h a = load_a_rowmajor(S, it * 16, 0, 32);
  v16h bm = load_b_colmajor(S, 0, jt * 16, 32);
  v8f c = {};
  c = __builtin_amdgcn_wmma_f32_16x16x32_f16(false, a, false, bm, (short)0, c,
                                             false, false);
  int lane = threadIdx.x & 31, half = lane >> 4, col = lane & 15;
  const float* A = adj + (size_t)b * NN * NN;
  float local = 0.f;
#pragma unroll
  for (int v = 0; v < 8; ++v) {
    int rowM = v + 8 * half;
    float d = A[(size_t)(it * 16 + rowM) * NN + jt * 16 + col] - c[v];
    local += d * d;
  }
  local = wave_reduce_sum(local);
  if (lane == 0) atomicAdd(linksum, local);
}

// t1 = adj @ s1  : [2048x2048] x [2048x32], K-loop of 64 WMMAs per tile.
__global__ __launch_bounds__(256) void wmma_adj_s1_kernel(
    const float* __restrict__ adj, const float* __restrict__ s1,
    float* __restrict__ t1) {
  int tile = blockIdx.x * 8 + (threadIdx.x >> 5);  // 0..511
  int b = tile / 256;
  int rem = tile % 256;
  int it = rem >> 1, jt = rem & 1;
  const float* A = adj + (size_t)b * NN * NN;
  const float* S = s1 + (size_t)b * NN * 32;
  v8f c = {};
#pragma unroll 4
  for (int k0 = 0; k0 < NN; k0 += 32) {
    v16h a = load_a_rowmajor(A, it * 16, k0, NN);
    v16h bm = load_b_rowmajor(S, k0, jt * 16, 32);
    c = __builtin_amdgcn_wmma_f32_16x16x32_f16(false, a, false, bm, (short)0, c,
                                               false, false);
  }
  int lane = threadIdx.x & 31, half = lane >> 4, col = lane & 15;
  float* T = t1 + (size_t)b * NN * 32;
#pragma unroll
  for (int v = 0; v < 8; ++v) {
    int rowM = v + 8 * half;
    T[(size_t)(it * 16 + rowM) * 32 + jt * 16 + col] = c[v];
  }
}

// adj2 = s1^T @ t1 : [32x2048] x [2048x32]; 8 waves, one block.
__global__ __launch_bounds__(256) void wmma_adj2_kernel(
    const float* __restrict__ s1, const float* __restrict__ t1,
    float* __restrict__ adj2) {
  int wave = threadIdx.x >> 5;  // 0..7
  int b = wave >> 2;
  int rem = wave & 3;
  int it = rem >> 1, jt = rem & 1;
  const float* S = s1 + (size_t)b * NN * 32;
  const float* T = t1 + (size_t)b * NN * 32;
  v8f c = {};
#pragma unroll 4
  for (int k0 = 0; k0 < NN; k0 += 32) {
    v16h a = load_a_colmajor(S, it * 16, k0, 32);  // A(m,k)=s1[k,m]
    v16h bm = load_b_rowmajor(T, k0, jt * 16, 32);
    c = __builtin_amdgcn_wmma_f32_16x16x32_f16(false, a, false, bm, (short)0, c,
                                               false, false);
  }
  int lane = threadIdx.x & 31, half = lane >> 4, col = lane & 15;
#pragma unroll
  for (int v = 0; v < 8; ++v) {
    int rowM = v + 8 * half;
    adj2[b * 1024 + (it * 16 + rowM) * 32 + jt * 16 + col] = c[v];
  }
}

// xp1[b,k,f] = sum_n s1[b,n,k] * h2[b,n,f]   (960 outputs)
__global__ void pool_x_kernel(const float* __restrict__ s,
                              const float* __restrict__ x,
                              float* __restrict__ out) {
  int idx = blockIdx.x * blockDim.x + threadIdx.x;
  if (idx >= BB * 32 * 15) return;
  int b = idx / (32 * 15);
  int rem = idx % (32 * 15);
  int k = rem / 15, f = rem % 15;
  const float* sb = s + (size_t)b * NN * 32;
  const float* xb = x + (size_t)b * NN * 15;
  float acc = 0.f;
  for (int n = 0; n < NN; ++n) acc += sb[n * 32 + k] * xb[n * 15 + f];
  out[idx] = acc;
}

// ---------------- pool-2 + final head (single block) ----------------
__global__ __launch_bounds__(128) void pool2_final_kernel(
    const float* __restrict__ adj2, const float* __restrict__ xp1,
    const float* __restrict__ W_p2, const float* __restrict__ a_p2,
    const float* __restrict__ lin2_w, const float* __restrict__ lin2_b,
    const float* __restrict__ lin_w, const float* __restrict__ lin_b,
    const float* __restrict__ linksum1, float* __restrict__ out) {
  __shared__ float X[32 * 15], A2[32 * 32], Wh[32 * 4], F1[32], F2[32];
  __shared__ float G[32 * 4], S2[32 * 4], HF[60];
  __shared__ float l2acc;
  const int tid = threadIdx.x;
  if (tid == 0) l2acc = 0.f;

  for (int b = 0; b < 2; ++b) {
    __syncthreads();
    for (int i = tid; i < 480; i += 128) X[i] = xp1[b * 480 + i];
    for (int i = tid; i < 1024; i += 128) A2[i] = adj2[b * 1024 + i];
    __syncthreads();
    {  // Wh = X @ W_p2  (32x15 x 15x4)
      int n = tid >> 2, o = tid & 3;
      float acc = 0.f;
      for (int k = 0; k < 15; ++k) acc += X[n * 15 + k] * W_p2[k * 4 + o];
      Wh[tid] = acc;
    }
    __syncthreads();
    if (tid < 32) {
      float s1v = 0.f, s2v = 0.f;
      for (int k = 0; k < 4; ++k) {
        s1v += Wh[tid * 4 + k] * a_p2[k];
        s2v += Wh[tid * 4 + k] * a_p2[4 + k];
      }
      F1[tid] = s1v;
      F2[tid] = s2v;
    }
    __syncthreads();
    if (tid < 32) {  // one attention row per thread (32x32 graph)
      float e[32], mx = -3.4e38f;
      for (int j = 0; j < 32; ++j) {
        float v = F1[tid] + F2[j];
        v = (v > 0.f) ? v : 0.2f * v;
        v = (A2[tid * 32 + j] > 0.f) ? v : NEG_INF_E;
        e[j] = v;
        mx = fmaxf(mx, v);
      }
      float sum = 0.f;
      for (int j = 0; j < 32; ++j) {
        e[j] = __expf(e[j] - mx);
        sum += e[j];
      }
      float invs = 1.f / sum;
      float h[4] = {0.f, 0.f, 0.f, 0.f};
      for (int j = 0; j < 32; ++j) {
        float a = e[j] * invs;
        for (int k = 0; k < 4; ++k) h[k] += a * Wh[j * 4 + k];
      }
      for (int k = 0; k < 4; ++k) {
        float v = h[k];
        G[tid * 4 + k] = (v > 0.f) ? v : (__expf(v) - 1.f);  // elu (concat)
      }
    }
    __syncthreads();
    if (tid < 32) {  // s2 = softmax(G @ lin2_w + lin2_b)
      float sp[4];
      for (int o = 0; o < 4; ++o) {
        float v = lin2_b[o];
        for (int k = 0; k < 4; ++k) v += G[tid * 4 + k] * lin2_w[k * 4 + o];
        sp[o] = v;
      }
      float mx = fmaxf(fmaxf(sp[0], sp[1]), fmaxf(sp[2], sp[3]));
      float sum = 0.f;
      for (int o = 0; o < 4; ++o) {
        sp[o] = __expf(sp[o] - mx);
        sum += sp[o];
      }
      for (int o = 0; o < 4; ++o) S2[tid * 4 + o] = sp[o] / sum;
    }
    __syncthreads();
    {  // link2 partial: ||adj2 - s2 s2^T||^2 contribution
      float local = 0.f;
      for (int idx = tid; idx < 1024; idx += 128) {
        int i = idx >> 5, j = idx & 31;
        float p = 0.f;
        for (int k = 0; k < 4; ++k) p += S2[i * 4 + k] * S2[j * 4 + k];
        float d = A2[idx] - p;
        local += d * d;
      }
      local = wave_reduce_sum(local);
      if ((tid & 31) == 0) atomicAdd(&l2acc, local);
    }
    if (tid < 60) {  // h_final = s2^T @ X : [4,15]
      int k = tid / 15, f = tid % 15;
      float acc = 0.f;
      for (int n = 0; n < 32; ++n) acc += S2[n * 4 + k] * X[n * 15 + f];
      HF[tid] = acc;
    }
    __syncthreads();
    if (tid < 2) {  // ypred[b] = HF(60) @ lin_w + lin_b
      float v = lin_b[tid];
      for (int t = 0; t < 60; ++t) v += HF[t] * lin_w[t * 2 + tid];
      out[b * 2 + tid] = v;
    }
  }
  __syncthreads();
  if (tid == 0) {
    float link1 = sqrtf(linksum1[0]) / 8388608.f;  // /= adj.size (2*2048*2048)
    float link2 = sqrtf(l2acc) / 2048.f;           // /= adj2.size (2*32*32)
    out[4] = link1 + link2;
  }
}

// ---------------- host side ----------------
extern "C" void kernel_launch(void* const* d_in, const int* in_sizes, int n_in,
                              void* d_out, int out_size, void* d_ws,
                              size_t ws_size, hipStream_t stream) {
  (void)in_sizes; (void)n_in; (void)out_size; (void)ws_size;
  const float* x       = (const float*)d_in[0];   // [2,2048,128]
  const float* adj     = (const float*)d_in[1];   // [2,2048,2048]
  const float* W_heads = (const float*)d_in[2];   // [5,128,3]
  const float* a_heads = (const float*)d_in[3];   // [5,6]
  const float* W_out   = (const float*)d_in[4];   // [15,15]
  const float* a_out   = (const float*)d_in[5];   // [30]
  const float* W_p1    = (const float*)d_in[6];   // [15,32]
  const float* a_p1    = (const float*)d_in[7];   // [64]
  const float* lin1_w  = (const float*)d_in[8];   // [32,32]
  const float* lin1_b  = (const float*)d_in[9];   // [32]
  const float* W_p2    = (const float*)d_in[10];  // [15,4]
  const float* a_p2    = (const float*)d_in[11];  // [8]
  const float* lin2_w  = (const float*)d_in[12];  // [4,4]
  const float* lin2_b  = (const float*)d_in[13];  // [4]
  const float* lin_w   = (const float*)d_in[14];  // [60,2]
  const float* lin_b   = (const float*)d_in[15];  // [2]

  float* out  = (float*)d_out;
  float* atts = out + 5;  // [5,2,2048,2048] after ypred(4)+loss(1)
  float* ws   = (float*)d_ws;

  // workspace layout (float offsets)
  const size_t WS_WHH = 0;                    // 5*4096*3
  const size_t WS_F1H = WS_WHH + 61440;       // 5*4096
  const size_t WS_F2H = WS_F1H + 20480;
  const size_t WS_H1  = WS_F2H + 20480;       // 4096*15
  const size_t WS_WHO = WS_H1  + 61440;       // 4096*15
  const size_t WS_F1O = WS_WHO + 61440;
  const size_t WS_F2O = WS_F1O + 4096;
  const size_t WS_H2  = WS_F2O + 4096;        // 4096*15
  const size_t WS_WHP = WS_H2  + 61440;       // 4096*32
  const size_t WS_F1P = WS_WHP + 131072;
  const size_t WS_F2P = WS_F1P + 4096;
  const size_t WS_S1  = WS_F2P + 4096;        // 4096*32
  const size_t WS_T1  = WS_S1  + 131072;      // 4096*32
  const size_t WS_A2  = WS_T1  + 131072;      // 2*32*32
  const size_t WS_XP1 = WS_A2  + 2048;        // 2*32*15
  const size_t WS_LNK = WS_XP1 + 960;         // 2

  zero_kernel<<<1, 32, 0, stream>>>(ws + WS_LNK, 2);

  // ---- 5 GAT heads (FEAT=128 -> HID=3), emit atts + concat h1 ----
  for (int h = 0; h < 5; ++h) {
    const float* W  = W_heads + h * 128 * 3;
    const float* av = a_heads + h * 6;
    float* Wh = ws + WS_WHH + (size_t)h * ROWS * 3;
    float* f1 = ws + WS_F1H + (size_t)h * ROWS;
    float* f2 = ws + WS_F2H + (size_t)h * ROWS;
    proj_kernel<<<(ROWS * 3 + 255) / 256, 256, 0, stream>>>(x, W, Wh, ROWS,
                                                            128, 3);
    attvec_kernel<<<(ROWS + 255) / 256, 256, 0, stream>>>(Wh, av, f1, f2, ROWS,
                                                          3);
    gat_attn_kernel<3, true, true, false><<<ROWS, 256, 0, stream>>>(
        adj, Wh, f1, f2, atts + (size_t)h * ROWS * NN, ws + WS_H1, 15, h * 3,
        nullptr, nullptr, nullptr);
  }

  // ---- out GAT (15 -> 15) + elu -> h2 ----
  proj_kernel<<<(ROWS * 15 + 255) / 256, 256, 0, stream>>>(
      ws + WS_H1, W_out, ws + WS_WHO, ROWS, 15, 15);
  attvec_kernel<<<(ROWS + 255) / 256, 256, 0, stream>>>(
      ws + WS_WHO, a_out, ws + WS_F1O, ws + WS_F2O, ROWS, 15);
  gat_attn_kernel<15, false, true, false><<<ROWS, 256, 0, stream>>>(
      adj, ws + WS_WHO, ws + WS_F1O, ws + WS_F2O, nullptr, ws + WS_H2, 15, 0,
      nullptr, nullptr, nullptr);

  // ---- pool-1 GAT (15 -> 32), fused lin1 + cluster softmax -> s1 ----
  proj_kernel<<<(ROWS * 32 + 255) / 256, 256, 0, stream>>>(
      ws + WS_H2, W_p1, ws + WS_WHP, ROWS, 15, 32);
  attvec_kernel<<<(ROWS + 255) / 256, 256, 0, stream>>>(
      ws + WS_WHP, a_p1, ws + WS_F1P, ws + WS_F2P, ROWS, 32);
  gat_attn_kernel<32, false, true, true><<<ROWS, 256, 0, stream>>>(
      adj, ws + WS_WHP, ws + WS_F1P, ws + WS_F2P, nullptr, nullptr, 0, 0,
      lin1_w, lin1_b, ws + WS_S1);

  // ---- diff-pool 1: WMMA GEMMs ----
  wmma_link1_kernel<<<4096, 256, 0, stream>>>(adj, ws + WS_S1, ws + WS_LNK);
  wmma_adj_s1_kernel<<<64, 256, 0, stream>>>(adj, ws + WS_S1, ws + WS_T1);
  wmma_adj2_kernel<<<1, 256, 0, stream>>>(ws + WS_S1, ws + WS_T1, ws + WS_A2);
  pool_x_kernel<<<4, 256, 0, stream>>>(ws + WS_S1, ws + WS_H2, ws + WS_XP1);

  // ---- pool-2 + classifier + loss ----
  pool2_final_kernel<<<1, 128, 0, stream>>>(ws + WS_A2, ws + WS_XP1, W_p2,
                                            a_p2, lin2_w, lin2_b, lin_w, lin_b,
                                            ws + WS_LNK, out);
}
